// MoELayer_85504208929089
// MI455X (gfx1250) — matrix operand order repaired and checked by reading, not compiled, not run on hardware
//
#include <hip/hip_runtime.h>
#include <hip/hip_bf16.h>

// ---------------------------------------------------------------------------
// Dense MoE: out[n,:] = sum_e softmax(x@Wg+bg)[n,e] * (gelu(x@W1[e]+b1[e])@W2[e]+b2[e])
// N=8192, D=1024, H=2048, O=1024, E=8.
// bf16 WMMA (v_wmma_f32_16x16x32_bf16), f32 accumulation.
// Block tile 256x128x32, 8 waves, each wave a 64x64 sub-tile (4x4 WMMA tiles).
// Double-buffered LDS; async global->LDS (gfx1250) when the builtin exists.
// ---------------------------------------------------------------------------

#define MOE_N 8192
#define MOE_D 1024
#define MOE_H 2048
#define MOE_O 1024
#define MOE_E 8

typedef unsigned short u16;
typedef __attribute__((ext_vector_type(16))) __bf16 v16bf;
typedef __attribute__((ext_vector_type(8)))  float  v8f;
typedef __attribute__((ext_vector_type(4)))  int    v4i;

union Frag {
    v16bf v;
    uint4 u[2];
};

#if defined(__gfx1250__) && __has_builtin(__builtin_amdgcn_global_load_async_to_lds_b128) && __has_builtin(__builtin_amdgcn_s_wait_asynccnt)
#define USE_ASYNC_LDS 1
typedef __attribute__((address_space(1))) v4i* gas_v4i;
typedef __attribute__((address_space(3))) v4i* las_v4i;
#else
#define USE_ASYNC_LDS 0
#endif

__device__ __forceinline__ u16 f2bf(float f) {
    unsigned u = __float_as_uint(f);
    unsigned r = (u + 0x7FFFu + ((u >> 16) & 1u)) >> 16;  // round-nearest-even
    return (u16)r;
}

// ---------------------------------------------------------------------------
// Gate: logits = x@Wg + bg ; softmax over E. One thread per token.
// ---------------------------------------------------------------------------
__global__ void gate_kernel(const float* __restrict__ x, const float* __restrict__ Wg,
                            const float* __restrict__ bg, float* __restrict__ gate) {
    int n = blockIdx.x * blockDim.x + threadIdx.x;
    if (n >= MOE_N) return;
    float acc[MOE_E];
#pragma unroll
    for (int e = 0; e < MOE_E; ++e) acc[e] = bg[e];
    const float* xr = x + (size_t)n * MOE_D;
    for (int d = 0; d < MOE_D; ++d) {
        float xv = xr[d];
        const float* wr = Wg + (size_t)d * MOE_E;
#pragma unroll
        for (int e = 0; e < MOE_E; ++e) acc[e] = fmaf(xv, wr[e], acc[e]);
    }
    float m = acc[0];
#pragma unroll
    for (int e = 1; e < MOE_E; ++e) m = fmaxf(m, acc[e]);
    float s = 0.0f;
#pragma unroll
    for (int e = 0; e < MOE_E; ++e) { acc[e] = __expf(acc[e] - m); s += acc[e]; }
    float inv = 1.0f / s;
    float* gr = gate + (size_t)n * MOE_E;
#pragma unroll
    for (int e = 0; e < MOE_E; ++e) gr[e] = acc[e] * inv;
}

// ---------------------------------------------------------------------------
// Elementwise f32 -> bf16 (x conversion)
// ---------------------------------------------------------------------------
__global__ void cvt_f32_bf16(const float* __restrict__ in, u16* __restrict__ out, size_t count) {
    size_t i = (size_t)blockIdx.x * blockDim.x + threadIdx.x;
    size_t stride = (size_t)gridDim.x * blockDim.x;
    for (; i < count; i += stride) out[i] = f2bf(in[i]);
}

// ---------------------------------------------------------------------------
// Batched transpose + convert: in [batch][R][C] f32 -> out [batch][C][R] bf16.
// ---------------------------------------------------------------------------
__global__ void transpose_cvt(const float* __restrict__ in, u16* __restrict__ out,
                              int R, int C) {
    __shared__ float tile[32][33];
    int c0 = blockIdx.x * 32;
    int r0 = blockIdx.y * 32;
    const float* inb = in + (size_t)blockIdx.z * R * C;
    u16* outb = out + (size_t)blockIdx.z * R * C;
    int tx = threadIdx.x;  // 0..31
    int ty = threadIdx.y;  // 0..7
#pragma unroll
    for (int i = 0; i < 32; i += 8)
        tile[ty + i][tx] = inb[(size_t)(r0 + ty + i) * C + (c0 + tx)];
    __syncthreads();
#pragma unroll
    for (int i = 0; i < 32; i += 8)
        outb[(size_t)(c0 + ty + i) * R + (r0 + tx)] = f2bf(tile[tx][ty + i]);
}

// ---------------------------------------------------------------------------
// bf16 WMMA GEMM: C[M][Nd] = A[M][K] @ Bt[Nd][K]^T   (both operands K-contiguous)
// EPI=0: +bias, gelu, bf16 store. EPI=1: gated f32 store/accumulate.
// ---------------------------------------------------------------------------
#define BM 256
#define BN 128
#define BK 32
#define LDSK 40  // padded LDS row stride in bf16 elems (80B = 20 banks, conflict-free)

// Stage one (A,B) K-slice into an LDS buffer. 256 rows(A)+128 rows(B), 4 x 16B
// chunks per row = 1536 chunks; 6 per thread. Chunk->A/B split is uniform in i.
__device__ __forceinline__ void stage_tiles(const u16* __restrict__ A,
                                            const u16* __restrict__ Bt, int K,
                                            int bm, int bn, int k0,
                                            u16* __restrict__ sAbuf,
                                            u16* __restrict__ sBbuf, int tid) {
#pragma unroll
    for (int i = 0; i < 6; ++i) {
        int c = tid + i * 256;
        const u16* gsrc;
        u16* ldst;
        if (i < 4) {  // c < 1024 -> A chunk
            int r = c >> 2, kc = (c & 3) * 8;
            gsrc = A + (size_t)(bm + r) * K + (k0 + kc);
            ldst = sAbuf + r * LDSK + kc;
        } else {      // B chunk
            int c2 = c - 1024;
            int r = c2 >> 2, kc = (c2 & 3) * 8;
            gsrc = Bt + (size_t)(bn + r) * K + (k0 + kc);
            ldst = sBbuf + r * LDSK + kc;
        }
#if USE_ASYNC_LDS
        __builtin_amdgcn_global_load_async_to_lds_b128(
            (gas_v4i)(const void*)gsrc, (las_v4i)(void*)ldst, 0, 0);
#else
        *(uint4*)ldst = *(const uint4*)gsrc;
#endif
    }
}

__device__ __forceinline__ void stage_fence() {
#if USE_ASYNC_LDS
    __builtin_amdgcn_s_wait_asynccnt(0);
#endif
    __syncthreads();
}

template <int EPI>
__global__ __launch_bounds__(256)
void gemm_bf16(const u16* __restrict__ A, const u16* __restrict__ Bt,
               const float* __restrict__ bias, int M, int Nd, int K,
               u16* __restrict__ outH,                       // EPI==0
               float* __restrict__ outF,                     // EPI==1
               const float* __restrict__ gate, int gstride,  // EPI==1
               int accumulate) {
    __shared__ u16 sA[2][BM * LDSK];  // 2 x 20 KB
    __shared__ u16 sB[2][BN * LDSK];  // 2 x 10 KB

    const int tid  = threadIdx.x;
    const int bm   = blockIdx.y * BM;
    const int bn   = blockIdx.x * BN;
    const int wave = tid >> 5;
    const int lane = tid & 31;
    const int wm   = wave >> 1;  // 0..3  (64-row slice)
    const int wn   = wave & 1;   // 0..1  (64-col slice)
    const int lrow = lane & 15;
    const int lhi  = lane >> 4;  // lane-half

    v8f acc[4][4];
#pragma unroll
    for (int mt = 0; mt < 4; ++mt)
#pragma unroll
        for (int nt = 0; nt < 4; ++nt) {
            v8f z = {0.f, 0.f, 0.f, 0.f, 0.f, 0.f, 0.f, 0.f};
            acc[mt][nt] = z;
        }

    // Prologue: stage K-slice 0 into buffer 0.
    stage_tiles(A, Bt, K, bm, bn, 0, sA[0], sB[0], tid);
    stage_fence();

    int cur = 0;
    for (int k0 = 0; k0 < K; k0 += BK) {
        // Prefetch next K-slice into the other buffer (overlaps with compute).
        if (k0 + BK < K)
            stage_tiles(A, Bt, K, bm, bn, k0 + BK, sA[cur ^ 1], sB[cur ^ 1], tid);

        // Fragments per ISA 7.12.2:
        //  A (16x32): lane-half lh holds K = {8lh..8lh+7} and {16+8lh..16+8lh+7}
        //  B (32x16): lane-half lh holds K = {8lh..8lh+7} and {16+8lh..16+8lh+7}
        Frag af[4], bfr[4];
#pragma unroll
        for (int mt = 0; mt < 4; ++mt) {
            const u16* p = sA[cur] + (wm * 64 + mt * 16 + lrow) * LDSK + lhi * 8;
            af[mt].u[0] = *(const uint4*)p;
            af[mt].u[1] = *(const uint4*)(p + 16);
        }
#pragma unroll
        for (int nt = 0; nt < 4; ++nt) {
            const u16* p = sB[cur] + (wn * 64 + nt * 16 + lrow) * LDSK + lhi * 8;
            bfr[nt].u[0] = *(const uint4*)p;
            bfr[nt].u[1] = *(const uint4*)(p + 16);
        }

#pragma unroll
        for (int mt = 0; mt < 4; ++mt)
#pragma unroll
            for (int nt = 0; nt < 4; ++nt)
                acc[mt][nt] = __builtin_amdgcn_wmma_f32_16x16x32_bf16(
                    false, af[mt].v, false, bfr[nt].v, (short)0, acc[mt][nt],
                    false, false);

        stage_fence();  // next buffer staged + everyone done reading cur
        cur ^= 1;
    }

    // Epilogue. C/D layout: vgpr r -> M = r + 8*lane_half, N = lane&15.
#pragma unroll
    for (int mt = 0; mt < 4; ++mt) {
#pragma unroll
        for (int nt = 0; nt < 4; ++nt) {
            int colg = bn + wn * 64 + nt * 16 + lrow;
            float b = bias[colg];
#pragma unroll
            for (int r = 0; r < 8; ++r) {
                int rowg = bm + wm * 64 + mt * 16 + r + lhi * 8;
                float v = acc[mt][nt][r] + b;
                if (EPI == 0) {
                    // tanh-approx GELU (jax.nn.gelu default)
                    float t = 0.7978845608028654f * (v + 0.044715f * v * v * v);
                    float g = 0.5f * v * (1.0f + tanhf(t));
                    outH[(size_t)rowg * Nd + colg] = f2bf(g);
                } else {
                    float g = gate[(size_t)rowg * gstride];
                    float o = g * v;
                    size_t idx = (size_t)rowg * Nd + colg;
                    if (accumulate) outF[idx] += o;
                    else            outF[idx]  = o;
                }
            }
        }
    }
}

// ---------------------------------------------------------------------------
// Workspace layout (bytes, 256-aligned):
//   gate_probs : N*E*4   =    262,144
//   xb (bf16)  : N*D*2   = 16,777,216
//   W1T (bf16) : E*H*D*2 = 33,554,432   ([E][H][D], K-contiguous)
//   W2T (bf16) : E*O*H*2 = 33,554,432   ([E][O][H], K-contiguous)
//   hb (bf16)  : N*H*2   = 33,554,432   (per-expert, reused)
// ---------------------------------------------------------------------------
#define OFF_GATE 0
#define OFF_XB   (size_t)262144
#define OFF_W1T  (size_t)(262144 + 16777216)
#define OFF_W2T  (OFF_W1T + (size_t)33554432)
#define OFF_HB   (OFF_W2T + (size_t)33554432)

extern "C" void kernel_launch(void* const* d_in, const int* in_sizes, int n_in,
                              void* d_out, int out_size, void* d_ws, size_t ws_size,
                              hipStream_t stream) {
    const float* x  = (const float*)d_in[0];
    const float* W1 = (const float*)d_in[1];
    const float* b1 = (const float*)d_in[2];
    const float* W2 = (const float*)d_in[3];
    const float* b2 = (const float*)d_in[4];
    const float* Wg = (const float*)d_in[5];
    const float* bg = (const float*)d_in[6];
    float* out = (float*)d_out;

    char* ws = (char*)d_ws;
    float* gate = (float*)(ws + OFF_GATE);
    u16*   xb   = (u16*)(ws + OFF_XB);
    u16*   W1T  = (u16*)(ws + OFF_W1T);
    u16*   W2T  = (u16*)(ws + OFF_W2T);
    u16*   hb   = (u16*)(ws + OFF_HB);

    // 1) gate probs
    gate_kernel<<<MOE_N / 256, 256, 0, stream>>>(x, Wg, bg, gate);

    // 2) conversions (one-time, bandwidth-trivial)
    cvt_f32_bf16<<<2048, 256, 0, stream>>>(x, xb, (size_t)MOE_N * MOE_D);
    transpose_cvt<<<dim3(MOE_H / 32, MOE_D / 32, MOE_E), dim3(32, 8, 1), 0, stream>>>(
        W1, W1T, MOE_D, MOE_H);
    transpose_cvt<<<dim3(MOE_O / 32, MOE_H / 32, MOE_E), dim3(32, 8, 1), 0, stream>>>(
        W2, W2T, MOE_H, MOE_O);

    // 3) per-expert fused GEMM1(+gelu) -> GEMM2(+gate combine); serialized on stream
    for (int e = 0; e < MOE_E; ++e) {
        gemm_bf16<0><<<dim3(MOE_H / BN, MOE_N / BM), 256, 0, stream>>>(
            xb, W1T + (size_t)e * MOE_H * MOE_D, b1 + (size_t)e * MOE_H,
            MOE_N, MOE_H, MOE_D, hb, nullptr, nullptr, 0, 0);
        gemm_bf16<1><<<dim3(MOE_O / BN, MOE_N / BM), 256, 0, stream>>>(
            hb, W2T + (size_t)e * MOE_O * MOE_H, b2 + (size_t)e * MOE_O,
            MOE_N, MOE_O, MOE_H, nullptr, out, gate + e, MOE_E, e > 0);
    }
}
